// Experts_3504693313888
// MI455X (gfx1250) — compile-verified, matrix operand order checked by reading, not run on hardware
//
#include <hip/hip_runtime.h>
#include <math.h>

// ---- problem constants (match reference) ----
#define NEXP    8
#define DMODEL  1024
#define DFF     4096
#define BATCH   4
#define CAP     1024          // tokens per (b, e)
#define MTILE   32            // tokens per workgroup
#define FCHUNK  128           // F columns per fused chunk (one 16-wide tile per wave)
#define XSTR    (DMODEL + 8)  // LDS row stride in halves (bank-conflict pad)
#define HSTR    (FCHUNK + 8)

#define W_ELEMS   ((size_t)NEXP * DFF * DMODEL)           // 33,554,432 per weight tensor
#define WS_NEED   (2 * W_ELEMS * sizeof(unsigned short))  // 128 MB

typedef __attribute__((ext_vector_type(16))) __bf16 v16bf;
typedef __attribute__((ext_vector_type(8)))  float  v8f;

// pack two fp32 -> two bf16 (truncate) in one v_perm_b32
static __device__ __forceinline__ unsigned int pkbf(float lo, float hi) {
    return __builtin_amdgcn_perm(__float_as_uint(hi), __float_as_uint(lo), 0x07060302u);
}

// RNE fp32 -> bf16 (GELU -> LDS store only)
static __device__ __forceinline__ unsigned short f2bf(float f) {
    unsigned int u = __float_as_uint(f);
    u += 0x7FFFu + ((u >> 16) & 1u);
    return (unsigned short)(u >> 16);
}

// 16 fp32 (4 float4: K+0..7 and K+16..23 groups) -> one bf16 WMMA operand
static __device__ __forceinline__ v16bf cvt16(float4 a, float4 b, float4 c, float4 d) {
    union { v16bf v; unsigned int u[8]; } r;
    r.u[0] = pkbf(a.x, a.y); r.u[1] = pkbf(a.z, a.w);
    r.u[2] = pkbf(b.x, b.y); r.u[3] = pkbf(b.z, b.w);
    r.u[4] = pkbf(c.x, c.y); r.u[5] = pkbf(c.z, c.w);
    r.u[6] = pkbf(d.x, d.y); r.u[7] = pkbf(d.z, d.w);
    return r.v;
}

// 16 contiguous bf16 halves (pre-swizzled chunk) -> WMMA operand, two b128 loads
static __device__ __forceinline__ v16bf ld16(const unsigned short* p) {
    union { v16bf v; uint4 q[2]; } r;
    r.q[0] = *(const uint4*)(p);
    r.q[1] = *(const uint4*)(p + 8);
    return r.v;
}

// LDS operand: 8 halves at base, 8 halves at base+16 halves
static __device__ __forceinline__ v16bf ldsA(const unsigned short* base) {
    union { v16bf v; uint4 q[2]; } r;
    r.q[0] = *(const uint4*)(base);
    r.q[1] = *(const uint4*)(base + 16);
    return r.v;
}

static __device__ __forceinline__ v8f wmma_bf16(v16bf a, v16bf b, v8f c) {
    return __builtin_amdgcn_wmma_f32_16x16x32_bf16(false, a, false, b, (short)0, c, false, false);
}

static __device__ __forceinline__ float gelu_exact(float x) {
    return 0.5f * x * (1.0f + erff(x * 0.70710678118654752f));
}

// -------------------------------------------------------------------------
// Weight pre-conversion: fp32 row-major [R][K]  ->  bf16 swizzled chunks.
// Chunk c = (ftg * (K/32) + k) * 32 + lane holds 16 contiguous halves:
//   row  = ftg*16 + (lane&15)
//   K    = k*32 + (lane>>4)*8 + {0..7}  and  +16..+23
// -------------------------------------------------------------------------
__global__ void __launch_bounds__(256)
convert_weights(const float* __restrict__ w, unsigned short* __restrict__ dst,
                int K, int kcShift /* log2(K/32) */)
{
    const int gid  = blockIdx.x * 256 + threadIdx.x;
    const int lane = gid & 31;
    const int k    = (gid >> 5) & ((1 << kcShift) - 1);
    const int ftg  = gid >> (5 + kcShift);

    const int row  = ftg * 16 + (lane & 15);
    const float* src = w + (size_t)row * K + (k << 5) + ((lane >> 4) << 3);

    float4 f0 = *(const float4*)(src);
    float4 f1 = *(const float4*)(src + 4);
    float4 f2 = *(const float4*)(src + 16);
    float4 f3 = *(const float4*)(src + 20);

    uint4 o0 = { pkbf(f0.x, f0.y), pkbf(f0.z, f0.w), pkbf(f1.x, f1.y), pkbf(f1.z, f1.w) };
    uint4 o1 = { pkbf(f2.x, f2.y), pkbf(f2.z, f2.w), pkbf(f3.x, f3.y), pkbf(f3.z, f3.w) };

    uint4* d = (uint4*)dst + (size_t)gid * 2;
    d[0] = o0;
    d[1] = o1;
}

// -------------------------------------------------------------------------
// Fused per-expert FFN: X(32x1024) -> H chunk(32x128) -> GELU -> Y(32x1024)
// WSPATH=true: B operands are pre-swizzled bf16, software-pipelined 2 deep.
// -------------------------------------------------------------------------
template <bool WSPATH>
__global__ void __launch_bounds__(256, 2)
moe_ffn_fused(const float* __restrict__ x,
              const float* __restrict__ w1,
              const float* __restrict__ b1,
              const float* __restrict__ w2,
              const float* __restrict__ b2,
              const unsigned short* __restrict__ w1bf,
              const unsigned short* __restrict__ w2bf,
              float* __restrict__ out)
{
    extern __shared__ unsigned short smem[];
    unsigned short* Xs = smem;                   // [MTILE][XSTR] bf16 bits
    unsigned short* Hs = smem + MTILE * XSTR;    // [MTILE][HSTR] bf16 bits

    const int tid  = threadIdx.x;
    const int wave = tid >> 5;
    const int lane = tid & 31;
    const int l16  = lane & 15;
    const int kgrp = (lane >> 4) * 8;

    // expert-major block order -> one expert's bf16 weights (16 MB) stay in L2
    const int bx = blockIdx.x;
    const int e  = bx >> 7;
    const int r  = bx & 127;
    const int b  = r >> 5;
    const int ct = r & 31;
    const int tokbase = b * (NEXP * CAP) + e * CAP + ct * MTILE;

    const float* w1e = w1 + (size_t)e * DFF * DMODEL;
    const float* w2e = w2 + (size_t)e * DMODEL * DFF;
    const float* b1e = b1 + e * DFF;
    const float* b2e = b2 + e * DMODEL;

    // ---- stage X tile (32 x 1024) into LDS as bf16 ----
    {
        const float* xt = x + (size_t)tokbase * DMODEL;
        for (int i = tid * 4; i < MTILE * DMODEL; i += 256 * 4) {
            const int row = i >> 10;
            const int col = i & (DMODEL - 1);
            float4 f = *(const float4*)(xt + i);
            unsigned int* dst = (unsigned int*)&Xs[row * XSTR + col];
            dst[0] = pkbf(f.x, f.y);
            dst[1] = pkbf(f.z, f.w);
        }
    }
    __syncthreads();

    const v8f vzero = {0.f, 0.f, 0.f, 0.f, 0.f, 0.f, 0.f, 0.f};
    v8f yacc[2][8];
#pragma unroll
    for (int s = 0; s < 2; ++s)
#pragma unroll
        for (int t = 0; t < 8; ++t) yacc[s][t] = vzero;

    const int dbase = e * (DMODEL / 16) + wave * 8;      // w2 16-row d-tile base

#pragma unroll 1
    for (int fc = 0; fc < DFF / FCHUNK; ++fc) {
        const int fbase = fc * FCHUNK;
        const int frow  = fbase + wave * 16 + l16;        // this lane's w1 row / H column
        const int ftg   = e * (DFF / 16) + fc * 8 + wave; // global 16-row f-tile index

        // ---- phase 1: H tile (32 x 16 per wave) = X * w1^T, K = 1024 ----
        v8f h0 = vzero, h1 = vzero;
        if (WSPATH) {
            // chunk stride per k-step = 32 lanes * 16 halves = 512
            const unsigned short* bb =
                w1bf + ((size_t)ftg * (DMODEL / 32) * 32 + lane) * 16;
            v16bf Ba = ld16(bb);                          // k = 0 in flight
            v16bf Bb = ld16(bb + 512);                    // k = 1 in flight
#pragma unroll 4
            for (int k = 0; k < DMODEL / 32; ++k) {
                const int kn = (k + 2 < DMODEL / 32) ? k + 2 : DMODEL / 32 - 1;
                v16bf Bn = ld16(bb + (size_t)kn * 512);   // k+2 in flight
                const int acol = k * 32 + kgrp;
                v16bf A0 = ldsA(&Xs[l16 * XSTR + acol]);
                v16bf A1 = ldsA(&Xs[(16 + l16) * XSTR + acol]);
                h0 = wmma_bf16(A0, Ba, h0);
                h1 = wmma_bf16(A1, Ba, h1);
                Ba = Bb;
                Bb = Bn;
            }
        } else {
#pragma unroll 4
            for (int k = 0; k < DMODEL / 32; ++k) {
                const float4* p = (const float4*)(w1e + (size_t)frow * DMODEL + kgrp + k * 32);
                v16bf Bm = cvt16(p[0], p[1], p[4], p[5]);
                const int acol = k * 32 + kgrp;
                v16bf A0 = ldsA(&Xs[l16 * XSTR + acol]);
                v16bf A1 = ldsA(&Xs[(16 + l16) * XSTR + acol]);
                h0 = wmma_bf16(A0, Bm, h0);
                h1 = wmma_bf16(A1, Bm, h1);
            }
        }

        const float b1v = b1e[frow];
        __syncthreads();                                 // prior Hs readers done

        // bias + exact GELU, store bf16 H chunk to LDS (C-layout: VGPR v -> row)
#pragma unroll
        for (int v = 0; v < 8; ++v) {
            const int r0 = (lane >> 4) * 8 + v;
            Hs[r0 * HSTR + wave * 16 + l16]        = f2bf(gelu_exact(h0[v] + b1v));
            Hs[(16 + r0) * HSTR + wave * 16 + l16] = f2bf(gelu_exact(h1[v] + b1v));
        }
        __syncthreads();

        // prefetch next w1 chunk (global_prefetch_b8)
        if (fc + 1 < DFF / FCHUNK) {
            if (WSPATH)
                __builtin_prefetch((const void*)(w1bf + (((size_t)(ftg + 8) * (DMODEL / 32)) * 32 + lane) * 16), 0, 1);
            else
                __builtin_prefetch((const void*)(w1e + (size_t)(frow + FCHUNK) * DMODEL + kgrp), 0, 1);
        }

        // ---- phase 2: Y (32 x 128 per wave) += H_chunk * w2^T, K = 128 ----
        if (WSPATH) {
            // halves offset of B chunk for (t, kk): ((dbase+t)*(DFF/32) + fc*4+kk)*512 + lane*16
            const unsigned short* w2l = w2bf + lane * 16;
            auto boff = [&](int j) -> size_t {
                const int kk = j >> 3, t = j & 7;
                return ((size_t)(dbase + t) * (DFF / 32) + (fc * (FCHUNK / 32) + kk)) * 512;
            };
            v16bf Ba = ld16(w2l + boff(0));              // j = 0 in flight
            v16bf Bb = ld16(w2l + boff(1));              // j = 1 in flight
#pragma unroll
            for (int kk = 0; kk < FCHUNK / 32; ++kk) {
                const int acol = kk * 32 + kgrp;
                v16bf A0 = ldsA(&Hs[l16 * HSTR + acol]);
                v16bf A1 = ldsA(&Hs[(16 + l16) * HSTR + acol]);
#pragma unroll
                for (int t = 0; t < 8; ++t) {
                    const int j    = kk * 8 + t;
                    const int jmax = (FCHUNK / 32) * 8 - 1;
                    const int jn   = (j + 2 <= jmax) ? j + 2 : jmax;
                    v16bf Bn = ld16(w2l + boff(jn));     // j+2 in flight
                    yacc[0][t] = wmma_bf16(A0, Ba, yacc[0][t]);
                    yacc[1][t] = wmma_bf16(A1, Ba, yacc[1][t]);
                    Ba = Bb;
                    Bb = Bn;
                }
            }
        } else {
#pragma unroll
            for (int kk = 0; kk < FCHUNK / 32; ++kk) {
                const int acol = kk * 32 + kgrp;
                v16bf A0 = ldsA(&Hs[l16 * HSTR + acol]);
                v16bf A1 = ldsA(&Hs[(16 + l16) * HSTR + acol]);
#pragma unroll
                for (int t = 0; t < 8; ++t) {
                    const int d = wave * 128 + t * 16 + l16;
                    const float4* q = (const float4*)(w2e + (size_t)d * DFF + fbase + kk * 32 + kgrp);
                    v16bf Bm = cvt16(q[0], q[1], q[4], q[5]);
                    yacc[0][t] = wmma_bf16(A0, Bm, yacc[0][t]);
                    yacc[1][t] = wmma_bf16(A1, Bm, yacc[1][t]);
                }
            }
        }
    }

    // ---- epilogue: + b2, write fp32 Y ----
#pragma unroll
    for (int t = 0; t < 8; ++t) {
        const int dcol = wave * 128 + t * 16 + l16;
        const float bias = b2e[dcol];
#pragma unroll
        for (int v = 0; v < 8; ++v) {
            const int rr = (lane >> 4) * 8 + v;
            out[(size_t)(tokbase + rr) * DMODEL + dcol]      = yacc[0][t][v] + bias;
            out[(size_t)(tokbase + 16 + rr) * DMODEL + dcol] = yacc[1][t][v] + bias;
        }
    }
}

extern "C" void kernel_launch(void* const* d_in, const int* in_sizes, int n_in,
                              void* d_out, int out_size, void* d_ws, size_t ws_size,
                              hipStream_t stream) {
    (void)in_sizes; (void)n_in; (void)out_size;
    const float* x  = (const float*)d_in[0];
    const float* w1 = (const float*)d_in[1];
    const float* b1 = (const float*)d_in[2];
    const float* w2 = (const float*)d_in[3];
    const float* b2 = (const float*)d_in[4];
    float* out = (float*)d_out;

    const int grid = NEXP * BATCH * (CAP / MTILE);                  // 1024 workgroups
    const size_t shmem = (size_t)(MTILE * XSTR + MTILE * HSTR) * sizeof(unsigned short); // ~73 KB

    if (ws_size >= WS_NEED && d_ws != nullptr) {
        unsigned short* w1bf = (unsigned short*)d_ws;
        unsigned short* w2bf = w1bf + W_ELEMS;
        const int cgrid = (int)(W_ELEMS / 16 / 256);                // 8192 blocks
        convert_weights<<<dim3(cgrid), dim3(256), 0, stream>>>(w1, w1bf, DMODEL, 5);
        convert_weights<<<dim3(cgrid), dim3(256), 0, stream>>>(w2, w2bf, DFF, 7);
        moe_ffn_fused<true><<<dim3(grid), dim3(256), shmem, stream>>>(
            x, w1, b1, w2, b2, w1bf, w2bf, out);
    } else {
        moe_ffn_fused<false><<<dim3(grid), dim3(256), shmem, stream>>>(
            x, w1, b1, w2, b2, nullptr, nullptr, out);
    }
}
